// GPT_41197326303919
// MI455X (gfx1250) — matrix operand order repaired and checked by reading, not compiled, not run on hardware
//
#include <hip/hip_runtime.h>
#include <math.h>

// ---------------- problem constants ----------------
constexpr int CB = 4, CS = 1024, CD = 1024, CH = 16, CL = 4, CV = 32000;
constexpr int CDH = 64;           // head dim
constexpr int NTOK = CB * CS;     // 4096 rows

typedef __attribute__((ext_vector_type(16))) __bf16 v16bf;
typedef __attribute__((ext_vector_type(8)))  __bf16 v8bf;
typedef __attribute__((ext_vector_type(8)))  float  v8f;
typedef __attribute__((ext_vector_type(4)))  unsigned u32x4;
typedef __attribute__((ext_vector_type(4)))  int      i32x4;
typedef __attribute__((ext_vector_type(8)))  int      i32x8;

#ifndef __has_builtin
#define __has_builtin(x) 0
#endif
#if __has_builtin(__builtin_amdgcn_tensor_load_to_lds) && \
    __has_builtin(__builtin_amdgcn_s_wait_tensorcnt)
#define USE_TDM 1
#else
#define USE_TDM 0
#endif

static __device__ inline v16bf pack2(v8bf lo, v8bf hi) {
  v16bf r;
#pragma unroll
  for (int i = 0; i < 8; ++i) { r[i] = lo[i]; r[i + 8] = hi[i]; }
  return r;
}
// A-fragment: two 16B chunks at p[0..7] and p[16..23]
static __device__ inline v16bf load_a_frag(const __bf16* p) {
  v8bf lo = *(const v8bf*)p;
  v8bf hi = *(const v8bf*)(p + 16);
  return pack2(lo, hi);
}
// B-fragment: 16 contiguous bf16
static __device__ inline v16bf load_contig16(const __bf16* p) {
  v8bf lo = *(const v8bf*)p;
  v8bf hi = *(const v8bf*)(p + 8);
  return pack2(lo, hi);
}
static __device__ inline v8f vzero8() {
  v8f z;
#pragma unroll
  for (int i = 0; i < 8; ++i) z[i] = 0.0f;
  return z;
}
// low 32 bits of a generic pointer to __shared__ = LDS byte offset
static __device__ inline unsigned lds_off_u32(const void* p) {
  return (unsigned)(unsigned long long)(size_t)p;
}

#if USE_TDM
// TDM: DMA one [128 rows x 32 bf16] tile from global (row stride K elems) into
// LDS with 16B padding after every 64B row -> LDS row stride 80B (LDP=40).
static __device__ inline void tdm_load_tileA(const __bf16* gsrc, unsigned ldsoff,
                                             int Mtot, int Ktot) {
  unsigned long long ga = (unsigned long long)(size_t)gsrc;
  u32x4 g0 = {1u,                                   // count=1 (user descriptor)
              ldsoff,
              (unsigned)ga,
              (unsigned)((ga >> 32) & 0x01FFFFFFull) | (2u << 30)};  // type=2
  int w0 = (1 << 16)            // data_size = 1 (2 bytes)
         | (1 << 20)            // pad_enable
         | (3 << 22)            // pad_interval: code 3 = every 16 DWORDs (64B)
         | (3 << 25);           // pad_amount:  code 3 = 4 DWORDs (16B)
  i32x8 g1 = {w0,
              (Ktot & 0xFFFF) << 16,                              // tensor_dim0 lo
              ((Ktot >> 16) & 0xFFFF) | ((Mtot & 0xFFFF) << 16),  // dim0 hi | dim1 lo
              ((Mtot >> 16) & 0xFFFF) | (32 << 16),               // dim1 hi | tile_dim0
              128,                                                // tile_dim1
              Ktot,                                               // dim0_stride lo32
              0, 0};
  i32x4 z4 = {0, 0, 0, 0};
#if __clang_major__ >= 23
  i32x8 z8 = {0, 0, 0, 0, 0, 0, 0, 0};
  __builtin_amdgcn_tensor_load_to_lds(g0, g1, z4, z4, z8, 0);
#else
  __builtin_amdgcn_tensor_load_to_lds(g0, g1, z4, z4, 0);
#endif
}
#endif

// ---------------- state encoder ----------------
__global__ __launch_bounds__(256) void k_enc1(
    const float* __restrict__ states, const float* __restrict__ w1,
    const float* __restrict__ b1, const float* __restrict__ w2,
    const float* __restrict__ b2, float* __restrict__ h2out) {
  int tok = blockIdx.x * 256 + threadIdx.x;
  if (tok >= NTOK) return;
  float s0 = states[tok * 4 + 0], s1 = states[tok * 4 + 1];
  float s2 = states[tok * 4 + 2], s3 = states[tok * 4 + 3];
  float h1[16];
#pragma unroll
  for (int j = 0; j < 16; ++j) {
    float a = b1[j] + s0 * w1[0 * 16 + j] + s1 * w1[1 * 16 + j] +
              s2 * w1[2 * 16 + j] + s3 * w1[3 * 16 + j];
    h1[j] = fmaxf(a, 0.0f);
  }
#pragma unroll
  for (int j = 0; j < 16; ++j) {
    float a = b2[j];
#pragma unroll
    for (int k = 0; k < 16; ++k) a += h1[k] * w2[k * 16 + j];
    h2out[tok * 16 + j] = fmaxf(a, 0.0f);
  }
}

__global__ __launch_bounds__(256) void k_enc2(
    const float* __restrict__ h2, const float* __restrict__ w3,
    const float* __restrict__ b3, const float* __restrict__ pos_emb,
    const float* __restrict__ gpos, const int* __restrict__ tsteps,
    float* __restrict__ x) {
  int idx = blockIdx.x * 256 + threadIdx.x;  // NTOK*CD
  int tok = idx >> 10, d = idx & (CD - 1);
  int b = tok >> 10, s = tok & (CS - 1);
  float acc = b3[d];
#pragma unroll
  for (int k = 0; k < 16; ++k) acc += h2[tok * 16 + k] * w3[k * CD + d];
  int ts = tsteps[b];
  x[idx] = tanhf(acc) + gpos[(size_t)ts * CD + d] + pos_emb[(size_t)s * CD + d];
}

// ---------------- layernorm (f32 in, bf16 out) ----------------
__global__ __launch_bounds__(256) void k_ln(
    const float* __restrict__ x, const float* __restrict__ g,
    const float* __restrict__ b, __bf16* __restrict__ out) {
  int row = blockIdx.x;
  const float* xr = x + (size_t)row * CD;
  __shared__ float red[256];
  int t = threadIdx.x;
  float v[4];
  float s = 0.0f;
#pragma unroll
  for (int i = 0; i < 4; ++i) { v[i] = xr[t + i * 256]; s += v[i]; }
  red[t] = s; __syncthreads();
  for (int o = 128; o > 0; o >>= 1) { if (t < o) red[t] += red[t + o]; __syncthreads(); }
  float mu = red[0] * (1.0f / CD);
  __syncthreads();
  float s2 = 0.0f;
#pragma unroll
  for (int i = 0; i < 4; ++i) { float d0 = v[i] - mu; s2 += d0 * d0; }
  red[t] = s2; __syncthreads();
  for (int o = 128; o > 0; o >>= 1) { if (t < o) red[t] += red[t + o]; __syncthreads(); }
  float rstd = rsqrtf(red[0] * (1.0f / CD) + 1e-5f);
#pragma unroll
  for (int i = 0; i < 4; ++i) {
    int c = t + i * 256;
    out[(size_t)row * CD + c] = (__bf16)((v[i] - mu) * rstd * g[c] + b[c]);
  }
}

// ---------------- bf16 WMMA GEMM ----------------
// C[M,N] = act(A_bf16[M,K] @ B_f32[K,N] + bias) (+resid)
// Block tile 128 x (4*TN), 8 waves (2 x 4), wave tile 64 x TN.
// TN=32: 4x2 accs (occupancy-friendly, N=1024 GEMMs).
// TN=64: 4x4 accs (16 wmma/step, for MLP-up and LM head).
// Double-buffered LDS, one barrier/step; A tile via TDM (padded rows) when
// available; B tile: f32 loads -> packed bf16 pair -> ds_store_b32, transposed.
constexpr int BM = 128, BK = 32, LDP = 40;  // LDP elems = 80B LDS row

template <int TN, int ACT, bool BIASF, bool RESF, bool OUTFF, bool OUTBF>
__global__ __launch_bounds__(256) void k_gemm(
    const __bf16* __restrict__ A, const float* __restrict__ Bw,
    const float* __restrict__ bias, const float* __restrict__ resid,
    float* __restrict__ outF, __bf16* __restrict__ outB,
    int M, int N, int K) {
  constexpr int BN = 4 * TN;
  constexpr int NFRAG = TN / 16;
  constexpr int BCN = BN / 16;   // staged columns per thread
  __shared__ __bf16 sA[2][BM][LDP];
  __shared__ __bf16 sBt[2][BN][LDP];   // transposed [n][k]
  int n0 = blockIdx.x * BN, m0 = blockIdx.y * BM;
  int tid = threadIdx.x;
  int wave = tid >> 5, lane = tid & 31;
  int wm = (wave >> 2) * 64;
  int wn = (wave & 3) * TN;
  int lh = lane >> 4, ln = lane & 15;

  // B staging: thread -> k-pair (2*bkp, 2*bkp+1) x BCN columns
  int bkp = tid >> 4;
  int bnb = (tid & 15) * BCN;

  v8f acc[4][NFRAG];
#pragma unroll
  for (int mi = 0; mi < 4; ++mi)
#pragma unroll
    for (int ni = 0; ni < NFRAG; ++ni) acc[mi][ni] = vzero8();

  float br0[BCN], br1[BCN];
  {  // preload B tile 0 into regs
    const float* g0 = Bw + (size_t)(2 * bkp) * N + n0 + bnb;
    const float* g1 = g0 + N;
#pragma unroll
    for (int i = 0; i < BCN; ++i) { br0[i] = g0[i]; br1[i] = g1[i]; }
  }
#if USE_TDM
  if (tid < 32)  // wave 0 drives the Tensor Data Mover
    tdm_load_tileA(A + (size_t)m0 * K, lds_off_u32(&sA[0][0][0]), M, K);
#else
  int ar = tid >> 1, ac = (tid & 1) * 16;
  v8bf ar0, ar1;
  {
    const v8bf* gp = (const v8bf*)(A + (size_t)(m0 + ar) * K + ac);
    ar0 = gp[0]; ar1 = gp[1];
  }
#endif

  int nsteps = K / BK;
  for (int s = 0; s < nsteps; ++s) {
    int buf = s & 1;
    // stage B(s): pack 2 K-consecutive bf16 -> one b32 store, transposed
#pragma unroll
    for (int i = 0; i < BCN; ++i) {
      union { __bf16 h[2]; unsigned u; } pk;
      pk.h[0] = (__bf16)br0[i];
      pk.h[1] = (__bf16)br1[i];
      *(unsigned*)&sBt[buf][bnb + i][2 * bkp] = pk.u;
    }
#if USE_TDM
    if (tid < 32) __builtin_amdgcn_s_wait_tensorcnt(0);  // A(s) landed
#else
    *(v8bf*)&sA[buf][ar][ac]     = ar0;
    *(v8bf*)&sA[buf][ar][ac + 8] = ar1;
#endif
    __syncthreads();
    if (s + 1 < nsteps) {  // prefetch next tiles for the other buffer
      int k0n = (s + 1) * BK;
      const float* g0 = Bw + (size_t)(k0n + 2 * bkp) * N + n0 + bnb;
      const float* g1 = g0 + N;
#pragma unroll
      for (int i = 0; i < BCN; ++i) { br0[i] = g0[i]; br1[i] = g1[i]; }
#if USE_TDM
      if (tid < 32)
        tdm_load_tileA(A + (size_t)m0 * K + k0n,
                       lds_off_u32(&sA[buf ^ 1][0][0]), M, K);
#else
      const v8bf* gp = (const v8bf*)(A + (size_t)(m0 + ar) * K + k0n + ac);
      ar0 = gp[0]; ar1 = gp[1];
#endif
    }
    // fragments + wmma: hold 4 A-frags, stream B-frags
    v16bf af[4];
#pragma unroll
    for (int mi = 0; mi < 4; ++mi)
      af[mi] = load_a_frag(&sA[buf][wm + mi * 16 + ln][lh * 8]);
#pragma unroll
    for (int ni = 0; ni < NFRAG; ++ni) {
      v16bf bfr = load_a_frag(&sBt[buf][wn + ni * 16 + ln][lh * 8]);
#pragma unroll
      for (int mi = 0; mi < 4; ++mi)
        acc[mi][ni] = __builtin_amdgcn_wmma_f32_16x16x32_bf16(
            false, af[mi], false, bfr, (short)0, acc[mi][ni], false, false);
    }
  }

#pragma unroll
  for (int mi = 0; mi < 4; ++mi)
#pragma unroll
    for (int ni = 0; ni < NFRAG; ++ni) {
      int gcol = n0 + wn + ni * 16 + ln;
      float bv = BIASF ? bias[gcol] : 0.0f;
#pragma unroll
      for (int r = 0; r < 8; ++r) {
        int grow = m0 + wm + mi * 16 + r + 8 * lh;
        float v = acc[mi][ni][r] + bv;
        if (ACT == 1) v = 0.5f * v * (1.0f + erff(v * 0.70710678f));  // exact gelu
        size_t off = (size_t)grow * N + gcol;
        if (RESF) v += resid[off];
        if (OUTFF) outF[off] = v;
        if (OUTBF) outB[off] = (__bf16)v;
      }
    }
}

// ---------------- flash attention (bf16 WMMA, causal) ----------------
__global__ __launch_bounds__(256) void k_attn(
    const __bf16* __restrict__ q, const __bf16* __restrict__ k,
    const __bf16* __restrict__ v, __bf16* __restrict__ y) {
  int bh = blockIdx.x;
  int b = bh >> 4, h = bh & (CH - 1);
  int wave = threadIdx.x >> 5, lane = threadIdx.x & 31;
  int lh = lane >> 4, ln = lane & 15;
  int q0 = blockIdx.y * 128 + wave * 16;

  __shared__ float pbuf[8][16 * 32];
  float* pb = pbuf[wave];

  const __bf16* qbase = q + (size_t)b * CS * CD + h * CDH;
  const __bf16* kbase = k + (size_t)b * CS * CD + h * CDH;
  const __bf16* vbase = v + (size_t)b * CS * CD + h * CDH;

  v16bf qf[2];
#pragma unroll
  for (int f = 0; f < 2; ++f)
    qf[f] = load_a_frag(qbase + (size_t)(q0 + ln) * CD + f * 32 + lh * 8);

  v8f o[4];
#pragma unroll
  for (int t = 0; t < 4; ++t) o[t] = vzero8();
  float m[8], l[8];
#pragma unroll
  for (int r = 0; r < 8; ++r) { m[r] = -1e30f; l[r] = 0.0f; }

  const float scale = 0.125f;  // 1/sqrt(64)
  int kmax = q0 + 16;

  for (int j = 0; j < kmax; j += 32) {
    v8f st[2];
#pragma unroll
    for (int nt = 0; nt < 2; ++nt) {
      v8f c = vzero8();
#pragma unroll
      for (int f = 0; f < 2; ++f) {
        int key = j + nt * 16 + ln;
        if (key > CS - 1) key = CS - 1;  // address clamp; masked below
        v16bf kf = load_a_frag(kbase + (size_t)key * CD + f * 32 + lh * 8);
        c = __builtin_amdgcn_wmma_f32_16x16x32_bf16(
            false, qf[f], false, kf, (short)0, c, false, false);
      }
      st[nt] = c;
    }
#pragma unroll
    for (int r = 0; r < 8; ++r) {
      int row = q0 + r + 8 * lh;
      float s0 = st[0][r] * scale;
      float s1 = st[1][r] * scale;
      if (j + ln > row) s0 = -1e30f;
      if (j + 16 + ln > row) s1 = -1e30f;
      float mx = fmaxf(s0, s1);
#pragma unroll
      for (int d = 1; d < 16; d <<= 1) mx = fmaxf(mx, __shfl_xor(mx, d, 16));
      float nm = fmaxf(m[r], mx);
      float alpha = __expf(m[r] - nm);
      float p0 = __expf(s0 - nm);
      float p1 = __expf(s1 - nm);
      float ps = p0 + p1;
#pragma unroll
      for (int d = 1; d < 16; d <<= 1) ps += __shfl_xor(ps, d, 16);
      l[r] = l[r] * alpha + ps;
      m[r] = nm;
#pragma unroll
      for (int t = 0; t < 4; ++t) o[t][r] *= alpha;
      pb[(r + 8 * lh) * 32 + ln]      = p0;
      pb[(r + 8 * lh) * 32 + 16 + ln] = p1;
    }
    __builtin_amdgcn_wave_barrier();  // keep LDS store->load ordering
    v16bf pf;
    {
      const float* pp = pb + ln * 32 + lh * 8;
#pragma unroll
      for (int i = 0; i < 8; ++i) {
        pf[i]     = (__bf16)pp[i];
        pf[i + 8] = (__bf16)pp[i + 16];
      }
    }
#pragma unroll
    for (int t = 0; t < 4; ++t) {
      v16bf vf;
      int dhc = t * 16 + ln;
#pragma unroll
      for (int i = 0; i < 16; ++i) {
        int key = j + lh * 8 + (i < 8 ? i : i + 8);
        if (key > CS - 1) key = CS - 1;
        vf[i] = vbase[(size_t)key * CD + dhc];
      }
      o[t] = __builtin_amdgcn_wmma_f32_16x16x32_bf16(
          false, pf, false, vf, (short)0, o[t], false, false);
    }
  }
#pragma unroll
  for (int r = 0; r < 8; ++r) l[r] = 1.0f / l[r];
#pragma unroll
  for (int t = 0; t < 4; ++t)
#pragma unroll
    for (int r = 0; r < 8; ++r) {
      int row = q0 + r + 8 * lh;
      y[(size_t)b * CS * CD + (size_t)row * CD + h * CDH + t * 16 + ln] =
          (__bf16)(o[t][r] * l[r]);
    }
}

// ---------------- launcher ----------------
extern "C" void kernel_launch(void* const* d_in, const int* in_sizes, int n_in,
                              void* d_out, int out_size, void* d_ws, size_t ws_size,
                              hipStream_t stream) {
  const float* states  = (const float*)d_in[0];
  const int*   tsteps  = (const int*)d_in[1];
  const float* se_w1   = (const float*)d_in[2];
  const float* se_b1   = (const float*)d_in[3];
  const float* se_w2   = (const float*)d_in[4];
  const float* se_b2   = (const float*)d_in[5];
  const float* se_w3   = (const float*)d_in[6];
  const float* se_b3   = (const float*)d_in[7];
  const float* pos_emb = (const float*)d_in[8];
  const float* gpos    = (const float*)d_in[9];
  const float* ln1_g   = (const float*)d_in[10];
  const float* ln1_b   = (const float*)d_in[11];
  const float* Wq      = (const float*)d_in[12];
  const float* Wk      = (const float*)d_in[13];
  const float* Wv      = (const float*)d_in[14];
  const float* Wp      = (const float*)d_in[15];
  const float* bp      = (const float*)d_in[16];
  const float* ln2_g   = (const float*)d_in[17];
  const float* ln2_b   = (const float*)d_in[18];
  const float* W1      = (const float*)d_in[19];
  const float* b1      = (const float*)d_in[20];
  const float* W2      = (const float*)d_in[21];
  const float* b2      = (const float*)d_in[22];
  const float* lnf_g   = (const float*)d_in[23];
  const float* lnf_b   = (const float*)d_in[24];
  const float* head_w  = (const float*)d_in[25];

  char* ws = (char*)d_ws;
  size_t off = 0;
  auto carve = [&](size_t bytes) {
    char* p = ws + off;
    off += (bytes + 255) & ~(size_t)255;
    return p;
  };
  float*  x  = (float*)carve((size_t)NTOK * CD * 4);
  __bf16* hn = (__bf16*)carve((size_t)NTOK * CD * 2);
  __bf16* qb = (__bf16*)carve((size_t)NTOK * CD * 2);
  __bf16* kb = (__bf16*)carve((size_t)NTOK * CD * 2);
  __bf16* vb = (__bf16*)carve((size_t)NTOK * CD * 2);
  __bf16* yb = (__bf16*)carve((size_t)NTOK * CD * 2);
  __bf16* mh = (__bf16*)carve((size_t)NTOK * 4 * CD * 2);
  float*  h2 = (float*)carve((size_t)NTOK * 16 * 4);
  (void)ws_size;

  k_enc1<<<NTOK / 256, 256, 0, stream>>>(states, se_w1, se_b1, se_w2, se_b2, h2);
  k_enc2<<<(NTOK * CD) / 256, 256, 0, stream>>>(h2, se_w3, se_b3, pos_emb, gpos,
                                                tsteps, x);

  dim3 gProj(CD / 128, NTOK / 128);        // (8,32)   TN=32 tiles
  dim3 gMlp1(4 * CD / 256, NTOK / 128);    // (16,32)  TN=64 tiles
  dim3 gHead(CV / 256, NTOK / 128);        // (125,32) TN=64 tiles
  dim3 gAttn(CB * CH, CS / 128);           // (64,8)

  for (int lyr = 0; lyr < CL; ++lyr) {
    size_t dd = (size_t)lyr * CD * CD;
    size_t d1 = (size_t)lyr * CD;
    size_t d4 = (size_t)lyr * CD * 4 * CD;
    k_ln<<<NTOK, 256, 0, stream>>>(x, ln1_g + d1, ln1_b + d1, hn);
    k_gemm<32, 0, false, false, false, true><<<gProj, 256, 0, stream>>>(
        hn, Wq + dd, nullptr, nullptr, nullptr, qb, NTOK, CD, CD);
    k_gemm<32, 0, false, false, false, true><<<gProj, 256, 0, stream>>>(
        hn, Wk + dd, nullptr, nullptr, nullptr, kb, NTOK, CD, CD);
    k_gemm<32, 0, false, false, false, true><<<gProj, 256, 0, stream>>>(
        hn, Wv + dd, nullptr, nullptr, nullptr, vb, NTOK, CD, CD);
    k_attn<<<gAttn, 256, 0, stream>>>(qb, kb, vb, yb);
    k_gemm<32, 0, true, true, true, false><<<gProj, 256, 0, stream>>>(
        yb, Wp + dd, bp + d1, x, x, nullptr, NTOK, CD, CD);
    k_ln<<<NTOK, 256, 0, stream>>>(x, ln2_g + d1, ln2_b + d1, hn);
    k_gemm<64, 1, true, false, false, true><<<gMlp1, 256, 0, stream>>>(
        hn, W1 + d4, b1 + (size_t)lyr * 4 * CD, nullptr, nullptr, mh,
        NTOK, 4 * CD, CD);
    k_gemm<32, 0, true, true, true, false><<<gProj, 256, 0, stream>>>(
        mh, W2 + d4, b2 + d1, x, x, nullptr, NTOK, CD, 4 * CD);
  }
  k_ln<<<NTOK, 256, 0, stream>>>(x, lnf_g, lnf_b, hn);
  k_gemm<64, 0, false, false, true, false><<<gHead, 256, 0, stream>>>(
      hn, head_w, nullptr, nullptr, (float*)d_out, nullptr, NTOK, CV, CD);
}
